// LanguageModel_13632226198209
// MI455X (gfx1250) — matrix operand order repaired and checked by reading, not compiled, not run on hardware
//
#include <hip/hip_runtime.h>

// ---------------------------------------------------------------- constants
constexpr int V_ = 32000;
constexpr int E_ = 1024;
constexpr int B_ = 4;
constexpr int T_ = 1024;
constexpr int H_ = 16;
constexpr int L_ = 8;
constexpr int D_ = 64;
constexpr int BT = B_ * T_;          // 4096
constexpr float SCALE = 0.03125f;    // E^-0.5 = 1/32
constexpr float NEG_BIG = -3.0e38f;

typedef __bf16 bf16;
typedef __attribute__((ext_vector_type(16))) __bf16 v16bf;
typedef __attribute__((ext_vector_type(8)))  __bf16 bf16x8;
typedef __attribute__((ext_vector_type(8)))  float  v8f;
typedef __attribute__((ext_vector_type(4)))  unsigned v4u;
typedef __attribute__((ext_vector_type(8)))  int    v8i;
typedef __attribute__((ext_vector_type(4)))  int    v4i;

#if defined(__has_builtin)
#if __has_builtin(__builtin_amdgcn_tensor_load_to_lds) && __has_builtin(__builtin_amdgcn_s_wait_tensorcnt)
#define USE_TDM 1
#endif
#endif
#ifndef USE_TDM
#define USE_TDM 0
#endif

__device__ inline bf16 f2bf(float f) {
    unsigned u = __builtin_bit_cast(unsigned, f);
    unsigned r = u + 0x7FFFu + ((u >> 16) & 1u);   // round-to-nearest-even
    unsigned short h = (unsigned short)(r >> 16);
    return __builtin_bit_cast(bf16, h);
}

__device__ inline v16bf mkfrag(const bf16* p0, const bf16* p1) {
    bf16x8 lo = *(const bf16x8*)p0;
    bf16x8 hi = *(const bf16x8*)p1;
    return __builtin_shufflevector(lo, hi, 0,1,2,3,4,5,6,7,8,9,10,11,12,13,14,15);
}

__device__ inline v8f wmma_bf16(v16bf a, v16bf b, v8f c) {
    return __builtin_amdgcn_wmma_f32_16x16x32_bf16(
        /*neg_a*/false, a, /*neg_b*/false, b,
        /*c_mod*/(short)0, c, /*reuse_a*/false, /*reuse_b*/false);
}

#if USE_TDM
// Issue a 2D TDM tile load: tile_rows x tile_k bf16 elements, row stride = K_elems.
// gaddr = byte address of tile start; lds_addr = byte offset in LDS.
__device__ inline void tdm_load_2d(const void* gaddr, unsigned lds_addr,
                                   unsigned K_elems, unsigned rows_total,
                                   unsigned tile_k, unsigned tile_rows) {
    unsigned long long ga = (unsigned long long)gaddr;
    v4u g0;
    g0[0] = 1u;                                      // count = 1 valid descriptor
    g0[1] = lds_addr;                                // bits 63:32  lds_addr
    g0[2] = (unsigned)(ga & 0xffffffffu);            // bits 95:64  global_addr lo
    g0[3] = (unsigned)((ga >> 32) & 0x01ffffffu)     // bits 120:96 global_addr hi
          | 0x80000000u;                             // bits 127:126 type = 2
    v8i g1;
    g1[0] = 0x00010000;                              // data_size = 1 (2 bytes), mask=0
    g1[1] = (int)((K_elems & 0xffffu) << 16);        // tensor_dim0[15:0] @ 63:48
    g1[2] = (int)((K_elems >> 16) |
                  ((rows_total & 0xffffu) << 16));   // tensor_dim0 hi | tensor_dim1 lo
    g1[3] = (int)((rows_total >> 16) |
                  (tile_k << 16));                   // tensor_dim1 hi | tile_dim0
    g1[4] = (int)(tile_rows & 0xffffu);              // tile_dim1 (tile_dim2 = 0)
    g1[5] = (int)K_elems;                            // tensor_dim0_stride[31:0]
    g1[6] = 0;                                       // stride0 hi | stride1 lo
    g1[7] = 0;
    v4i z = {0, 0, 0, 0};
#if __clang_major__ >= 23
    v8i z8 = {0, 0, 0, 0, 0, 0, 0, 0};
    __builtin_amdgcn_tensor_load_to_lds(g0, g1, z, z, z8, 0);
#else
    __builtin_amdgcn_tensor_load_to_lds(g0, g1, z, z, 0);
#endif
}
__device__ inline unsigned lds_off(const void* p) {
    return (unsigned)(unsigned long long)p;          // low 32 bits = LDS address
}
#endif

// ---------------------------------------------------------------- embedding
__global__ void embed_kernel(const int* __restrict__ idx,
                             const float* __restrict__ tok,
                             const float* __restrict__ pos,
                             float* __restrict__ x) {
    long i = (long)blockIdx.x * blockDim.x + threadIdx.x;   // over BT*E
    if (i >= (long)BT * E_) return;
    int e  = (int)(i & (E_ - 1));
    int bt = (int)(i >> 10);           // E_ == 1024
    int t  = bt & (T_ - 1);
    x[i] = tok[(long)idx[bt] * E_ + e] + pos[(long)t * E_ + e];
}

// ---------------------------------------------------------------- layernorm -> bf16
__global__ __launch_bounds__(256)
void ln_bf16_kernel(const float* __restrict__ x,
                    const float* __restrict__ s,
                    const float* __restrict__ b,
                    bf16* __restrict__ out) {
    int row = blockIdx.x;
    const float* xr = x + (long)row * E_;
    __shared__ float rs[8], rq[8];
    float vals[4];
    float sum = 0.f, sq = 0.f;
    for (int j = 0; j < 4; ++j) {
        float v = xr[threadIdx.x + j * 256];
        vals[j] = v; sum += v; sq += v * v;
    }
    for (int off = 16; off; off >>= 1) {
        sum += __shfl_xor(sum, off, 32);
        sq  += __shfl_xor(sq,  off, 32);
    }
    int wave = threadIdx.x >> 5;
    if ((threadIdx.x & 31) == 0) { rs[wave] = sum; rq[wave] = sq; }
    __syncthreads();
    float ts = 0.f, tq = 0.f;
    for (int i = 0; i < 8; ++i) { ts += rs[i]; tq += rq[i]; }
    float mean = ts * (1.f / E_);
    float var  = tq * (1.f / E_) - mean * mean;
    float rstd = rsqrtf(var + 1e-5f);
    for (int j = 0; j < 4; ++j) {
        int c = threadIdx.x + j * 256;
        out[(long)row * E_ + c] = f2bf((vals[j] - mean) * rstd * s[c] + b[c]);
    }
}

// ---------------------------------------------------------------- weight converters
// generic transpose+convert: out[n*K + k] = in[k*N + n]
__global__ void transpose_bf16_kernel(const float* __restrict__ in,
                                      bf16* __restrict__ out,
                                      int K, int N) {
    long i = (long)blockIdx.x * blockDim.x + threadIdx.x;
    if (i >= (long)K * N) return;
    long k = i % K, n = i / K;
    out[i] = f2bf(in[k * N + n]);
}

// in: [H][E][D] per layer -> out[(h*D+d)*E + e]
__global__ void conv_qkvw_kernel(const float* __restrict__ in,
                                 bf16* __restrict__ out) {
    long i = (long)blockIdx.x * blockDim.x + threadIdx.x;   // over E*E
    if (i >= (long)E_ * E_) return;
    int e = (int)(i % E_);
    int n = (int)(i / E_);
    int h = n / D_, d = n % D_;
    out[i] = f2bf(in[((long)h * E_ + e) * D_ + d]);
}

// ---------------------------------------------------------------- WMMA GEMM
// C[M,N] = A[M,K](bf16, row-major) * Bt[N,K](bf16, B transposed)
// Block: 256 thr = 8 waves arranged 2(M) x 4(N); tile 128x128; K-step 32.
// Double-buffered LDS; tiles staged by the Tensor Data Mover when available.
__global__ __launch_bounds__(256)
void gemm_bf16_kernel(const bf16* __restrict__ A,
                      const bf16* __restrict__ Bt,
                      const float* __restrict__ bias,
                      const float* __restrict__ res,
                      float* __restrict__ outF,
                      bf16* __restrict__ outH,
                      int M, int N, int K, int relu) {
    __shared__ bf16 la[2][128 * 32];
    __shared__ bf16 lb[2][128 * 32];
    int n0 = blockIdx.x * 128, m0 = blockIdx.y * 128;
    int tid = threadIdx.x;
    int w = tid >> 5, lane = tid & 31;
    int half = lane >> 4, r16 = lane & 15;
    int wm = w >> 2, wn = w & 3;           // 2 x 4 wave grid

    v8f acc[4][2];
    for (int mt = 0; mt < 4; ++mt)
        for (int nt = 0; nt < 2; ++nt) acc[mt][nt] = (v8f)0.0f;

#if USE_TDM
    if (w == 0) {
        tdm_load_2d((const char*)A + ((long)m0 * K) * 2, lds_off(&la[0][0]),
                    (unsigned)K, (unsigned)M, 32u, 128u);
        tdm_load_2d((const char*)Bt + ((long)n0 * K) * 2, lds_off(&lb[0][0]),
                    (unsigned)K, (unsigned)N, 32u, 128u);
        __builtin_amdgcn_s_wait_tensorcnt(0);
    }
    __syncthreads();
    int p = 0;
    for (int k0 = 0; k0 < K; k0 += 32) {
        if (w == 0 && k0 + 32 < K) {
            tdm_load_2d((const char*)A + ((long)m0 * K + k0 + 32) * 2,
                        lds_off(&la[p ^ 1][0]), (unsigned)K, (unsigned)M, 32u, 128u);
            tdm_load_2d((const char*)Bt + ((long)n0 * K + k0 + 32) * 2,
                        lds_off(&lb[p ^ 1][0]), (unsigned)K, (unsigned)N, 32u, 128u);
        }
#else
    int srow = tid >> 1, scol = (tid & 1) * 16;
    {   // prologue stage into buffer 0
        *(v16bf*)&la[0][srow * 32 + scol] = *(const v16bf*)&A[(long)(m0 + srow) * K + scol];
        *(v16bf*)&lb[0][srow * 32 + scol] = *(const v16bf*)&Bt[(long)(n0 + srow) * K + scol];
    }
    __syncthreads();
    int p = 0;
    for (int k0 = 0; k0 < K; k0 += 32) {
        if (k0 + 32 < K) {
            *(v16bf*)&la[p ^ 1][srow * 32 + scol] =
                *(const v16bf*)&A[(long)(m0 + srow) * K + k0 + 32 + scol];
            *(v16bf*)&lb[p ^ 1][srow * 32 + scol] =
                *(const v16bf*)&Bt[(long)(n0 + srow) * K + k0 + 32 + scol];
        }
#endif
        // ---- compute from buffer p
        v16bf af[4];
        #pragma unroll
        for (int mt = 0; mt < 4; ++mt) {
            const bf16* pa = &la[p][(wm * 64 + mt * 16 + r16) * 32 + half * 16];
            af[mt] = mkfrag(pa, pa + 8);
        }
        #pragma unroll
        for (int nt = 0; nt < 2; ++nt) {
            v16bf bfr = *(const v16bf*)&lb[p][(wn * 32 + nt * 16 + r16) * 32 + half * 16];
            #pragma unroll
            for (int mt = 0; mt < 4; ++mt)
                acc[mt][nt] = wmma_bf16(af[mt], bfr, acc[mt][nt]);
        }
#if USE_TDM
        if (w == 0 && k0 + 32 < K) __builtin_amdgcn_s_wait_tensorcnt(0);
#endif
        __syncthreads();
        p ^= 1;
    }

    for (int nt = 0; nt < 2; ++nt) {
        int n = n0 + wn * 32 + nt * 16 + r16;
        float bi = bias ? bias[n] : 0.f;
        #pragma unroll
        for (int mt = 0; mt < 4; ++mt) {
            #pragma unroll
            for (int j = 0; j < 8; ++j) {
                int m = m0 + wm * 64 + mt * 16 + j + 8 * half;
                long o = (long)m * N + n;
                float v = acc[mt][nt][j] + bi;
                if (res)  v += res[o];
                if (relu) v = fmaxf(v, 0.f);
                if (outF) outF[o] = v;
                else      outH[o] = f2bf(v);
            }
        }
    }
}

// ---------------------------------------------------------------- flash attention
// q,k,v,out: bf16 [B,T,H,D] (== [BT][E] row-major with col = h*D+d)
__global__ __launch_bounds__(128)
void attn_kernel(const bf16* __restrict__ q,
                 const bf16* __restrict__ k,
                 const bf16* __restrict__ v,
                 bf16* __restrict__ out) {
    __shared__ bf16 lp[4][16][32];      // per-wave P tile (16 q x 32 k)
    __shared__ bf16 lvt[4][64][48];     // per-wave V^T tile (64 d x 32 k, padded)
    int w = threadIdx.x >> 5, lane = threadIdx.x & 31;
    int half = lane >> 4, r16 = lane & 15;
    int qt = blockIdx.x, h = blockIdx.y, b = blockIdx.z;
    int qbase = qt * 64 + w * 16;
    long base = ((long)b * T_) * E_ + (long)h * D_;

    // Q fragments (d = 0..31 and 32..63)
    const bf16* qp = q + base + (long)(qbase + r16) * E_;
    v16bf aq0 = mkfrag(qp + half * 16,      qp + 8 + half * 16);
    v16bf aq1 = mkfrag(qp + 32 + half * 16, qp + 40 + half * 16);

    v8f acc[4];
    for (int nt = 0; nt < 4; ++nt) acc[nt] = (v8f)0.0f;
    float mrow[8], lrow[8];
    for (int j = 0; j < 8; ++j) { mrow[j] = NEG_BIG; lrow[j] = 0.f; }

    int qend = qbase + 15;
    for (int kv0 = 0; kv0 <= qend; kv0 += 32) {
        // stage V^T into LDS: lvt[d][key]
        {
            const bf16* vp = v + base + (long)(kv0 + lane) * E_;
            #pragma unroll
            for (int d = 0; d < 64; d += 8) {
                bf16x8 ch = *(const bf16x8*)(vp + d);
                #pragma unroll
                for (int jj = 0; jj < 8; ++jj) lvt[w][d + jj][lane] = ch[jj];
            }
        }
        // scores: two 16-key tiles
        v8f s[2];
        #pragma unroll
        for (int tile = 0; tile < 2; ++tile) {
            const bf16* kp = k + base + (long)(kv0 + tile * 16 + r16) * E_;
            v16bf bk0 = *(const v16bf*)(kp + half * 16);
            v16bf bk1 = *(const v16bf*)(kp + 32 + half * 16);
            v8f c = (v8f)0.0f;
            c = wmma_bf16(aq0, bk0, c);
            c = wmma_bf16(aq1, bk1, c);
            s[tile] = c;
        }
        // scale + causal mask
        #pragma unroll
        for (int tile = 0; tile < 2; ++tile)
            #pragma unroll
            for (int j = 0; j < 8; ++j) {
                int qrow = qbase + j + 8 * half;
                int kcol = kv0 + tile * 16 + r16;
                float val = s[tile][j] * SCALE;
                s[tile][j] = (kcol <= qrow) ? val : NEG_BIG;
            }
        // online softmax update (row = j + 8*half; reduce across 16-lane halves)
        #pragma unroll
        for (int j = 0; j < 8; ++j) {
            float mt = fmaxf(s[0][j], s[1][j]);
            for (int off = 8; off; off >>= 1) mt = fmaxf(mt, __shfl_xor(mt, off, 16));
            float mnew  = fmaxf(mrow[j], mt);
            float alpha = __expf(mrow[j] - mnew);
            float p0 = __expf(s[0][j] - mnew);
            float p1 = __expf(s[1][j] - mnew);
            float ps = p0 + p1;
            for (int off = 8; off; off >>= 1) ps += __shfl_xor(ps, off, 16);
            lrow[j] = lrow[j] * alpha + ps;
            mrow[j] = mnew;
            #pragma unroll
            for (int nt = 0; nt < 4; ++nt) acc[nt][j] *= alpha;
            int row = j + 8 * half;
            lp[w][row][r16]      = f2bf(p0);
            lp[w][row][16 + r16] = f2bf(p1);
        }
        asm volatile("s_wait_dscnt 0" ::: "memory");   // wave-local LDS handoff
        // P x V
        const bf16* pp = &lp[w][r16][0];
        v16bf ap = mkfrag(pp + half * 16, pp + 8 + half * 16);
        #pragma unroll
        for (int nt = 0; nt < 4; ++nt) {
            v16bf bv = *(const v16bf*)&lvt[w][nt * 16 + r16][half * 16];
            acc[nt] = wmma_bf16(ap, bv, acc[nt]);
        }
    }
    // normalize and write out
    #pragma unroll
    for (int j = 0; j < 8; ++j) {
        float inv = 1.f / lrow[j];
        int qrow = qbase + j + 8 * half;
        bf16* op = out + base + (long)qrow * E_;
        #pragma unroll
        for (int nt = 0; nt < 4; ++nt)
            op[nt * 16 + r16] = f2bf(acc[nt][j] * inv);
    }
}

// ---------------------------------------------------------------- loss
__global__ __launch_bounds__(256)
void rowloss_kernel(const float* __restrict__ logits,
                    const int* __restrict__ targets,
                    float* __restrict__ partial) {
    int row = blockIdx.x;
    const float* lr = logits + (long)row * V_;
    __shared__ float red[8];
    float mx = NEG_BIG;
    for (int c = threadIdx.x; c < V_; c += 256) mx = fmaxf(mx, lr[c]);
    for (int off = 16; off; off >>= 1) mx = fmaxf(mx, __shfl_xor(mx, off, 32));
    if ((threadIdx.x & 31) == 0) red[threadIdx.x >> 5] = mx;
    __syncthreads();
    float m = red[0];
    for (int i = 1; i < 8; ++i) m = fmaxf(m, red[i]);
    __syncthreads();
    float se = 0.f;
    for (int c = threadIdx.x; c < V_; c += 256) se += __expf(lr[c] - m);
    for (int off = 16; off; off >>= 1) se += __shfl_xor(se, off, 32);
    if ((threadIdx.x & 31) == 0) red[threadIdx.x >> 5] = se;
    __syncthreads();
    float tot = 0.f;
    for (int i = 0; i < 8; ++i) tot += red[i];
    if (threadIdx.x == 0) {
        float lt = lr[targets[row]];
        partial[row] = -(lt - m - __logf(tot));
    }
}

__global__ __launch_bounds__(256)
void finloss_kernel(const float* __restrict__ partial, float* __restrict__ out) {
    __shared__ float red[8];
    float s = 0.f;
    for (int i = threadIdx.x; i < BT; i += 256) s += partial[i];
    for (int off = 16; off; off >>= 1) s += __shfl_xor(s, off, 32);
    if ((threadIdx.x & 31) == 0) red[threadIdx.x >> 5] = s;
    __syncthreads();
    if (threadIdx.x == 0) {
        float t = 0.f;
        for (int i = 0; i < 8; ++i) t += red[i];
        out[0] = t / (float)BT;
    }
}

// ---------------------------------------------------------------- host
extern "C" void kernel_launch(void* const* d_in, const int* in_sizes, int n_in,
                              void* d_out, int out_size, void* d_ws, size_t ws_size,
                              hipStream_t stream) {
    const int*   idx     = (const int*)d_in[0];
    const int*   targets = (const int*)d_in[1];
    const float* tok_emb = (const float*)d_in[2];
    const float* pos_emb = (const float*)d_in[3];
    const float* Wq   = (const float*)d_in[4];
    const float* Wk   = (const float*)d_in[5];
    const float* Wv   = (const float*)d_in[6];
    const float* Wp   = (const float*)d_in[7];
    const float* bp   = (const float*)d_in[8];
    const float* ln1s = (const float*)d_in[9];
    const float* ln1b = (const float*)d_in[10];
    const float* ln2s = (const float*)d_in[11];
    const float* ln2b = (const float*)d_in[12];
    const float* W1   = (const float*)d_in[13];
    const float* b1   = (const float*)d_in[14];
    const float* W2   = (const float*)d_in[15];
    const float* b2   = (const float*)d_in[16];
    const float* lnfs = (const float*)d_in[17];
    const float* lnfb = (const float*)d_in[18];
    const float* Wlm  = (const float*)d_in[19];

    float* logits = (float*)d_out;                          // [BT][V]
    float* lossp  = (float*)d_out + (long)BT * V_;          // scalar loss slot

    // workspace carve-up (256B aligned)
    char* ws = (char*)d_ws;
    size_t off = 0;
    auto carve = [&](size_t bytes) { char* p = ws + off; off = (off + bytes + 255) & ~(size_t)255; return p; };
    float* X    = (float*)carve((size_t)BT * E_ * 4);
    bf16*  Hb   = (bf16*) carve((size_t)BT * E_ * 2);
    bf16*  Qb   = (bf16*) carve((size_t)BT * E_ * 2);
    bf16*  Kb   = (bf16*) carve((size_t)BT * E_ * 2);
    bf16*  Vb   = (bf16*) carve((size_t)BT * E_ * 2);
    bf16*  ATT  = (bf16*) carve((size_t)BT * E_ * 2);
    bf16*  MID  = (bf16*) carve((size_t)BT * 4 * E_ * 2);
    bf16*  WQT  = (bf16*) carve((size_t)E_ * E_ * 2);
    bf16*  WKT  = (bf16*) carve((size_t)E_ * E_ * 2);
    bf16*  WVT  = (bf16*) carve((size_t)E_ * E_ * 2);
    bf16*  WPT  = (bf16*) carve((size_t)E_ * E_ * 2);
    bf16*  W1T  = (bf16*) carve((size_t)E_ * 4 * E_ * 2);
    bf16*  W2T  = (bf16*) carve((size_t)E_ * 4 * E_ * 2);
    bf16*  WLMT = (bf16*) carve((size_t)V_ * E_ * 2);
    float* LP   = (float*)carve((size_t)BT * 4);

    auto gemm = [&](const bf16* A, const bf16* Bt, const float* bias,
                    const float* res, float* oF, bf16* oH,
                    int M, int N, int K, int relu) {
        dim3 g(N / 128, M / 128);
        gemm_bf16_kernel<<<g, 256, 0, stream>>>(A, Bt, bias, res, oF, oH, M, N, K, relu);
    };

    // embedding
    embed_kernel<<<(BT * E_) / 256, 256, 0, stream>>>(idx, tok_emb, pos_emb, X);

    // lm-head weight once
    transpose_bf16_kernel<<<((long)E_ * V_) / 256, 256, 0, stream>>>(Wlm, WLMT, E_, V_);

    for (int l = 0; l < L_; ++l) {
        const long HED = (long)H_ * E_ * D_;   // == E*E
        conv_qkvw_kernel<<<(E_ * E_) / 256, 256, 0, stream>>>(Wq + l * HED, WQT);
        conv_qkvw_kernel<<<(E_ * E_) / 256, 256, 0, stream>>>(Wk + l * HED, WKT);
        conv_qkvw_kernel<<<(E_ * E_) / 256, 256, 0, stream>>>(Wv + l * HED, WVT);
        transpose_bf16_kernel<<<((long)E_ * E_) / 256, 256, 0, stream>>>(Wp + (long)l * E_ * E_, WPT, E_, E_);
        transpose_bf16_kernel<<<((long)E_ * 4 * E_) / 256, 256, 0, stream>>>(W1 + (long)l * E_ * 4 * E_, W1T, E_, 4 * E_);
        transpose_bf16_kernel<<<((long)E_ * 4 * E_) / 256, 256, 0, stream>>>(W2 + (long)l * 4 * E_ * E_, W2T, 4 * E_, E_);

        ln_bf16_kernel<<<BT, 256, 0, stream>>>(X, ln1s + l * E_, ln1b + l * E_, Hb);
        gemm(Hb, WQT, nullptr, nullptr, nullptr, Qb, BT, E_, E_, 0);
        gemm(Hb, WKT, nullptr, nullptr, nullptr, Kb, BT, E_, E_, 0);
        gemm(Hb, WVT, nullptr, nullptr, nullptr, Vb, BT, E_, E_, 0);

        attn_kernel<<<dim3(T_ / 64, H_, B_), 128, 0, stream>>>(Qb, Kb, Vb, ATT);

        gemm(ATT, WPT, bp + l * E_, X, X, nullptr, BT, E_, E_, 0);

        ln_bf16_kernel<<<BT, 256, 0, stream>>>(X, ln2s + l * E_, ln2b + l * E_, Hb);
        gemm(Hb, W1T, b1 + l * 4 * E_, nullptr, nullptr, MID, BT, 4 * E_, E_, 1);
        gemm(MID, W2T, b2 + l * E_, X, X, nullptr, BT, E_, 4 * E_, 0);
    }

    ln_bf16_kernel<<<BT, 256, 0, stream>>>(X, lnfs, lnfb, Hb);
    gemm(Hb, WLMT, nullptr, nullptr, logits, nullptr, BT, V_, E_, 0);

    rowloss_kernel<<<BT, 256, 0, stream>>>(logits, targets, LP);
    finloss_kernel<<<1, 256, 0, stream>>>(LP, lossp);
}
